// NoisyMixtureOfExperts_71536975282232
// MI455X (gfx1250) — compile-verified, matrix operand order checked by reading, not run on hardware
//
#include <hip/hip_runtime.h>
#include <hip/hip_bf16.h>
#include <math.h>

#define DIM   768
#define NEXP  1536
#define EDIM  64
#define NTOK  4096
#define NOISE_SCALE 0.1f

typedef __attribute__((ext_vector_type(16))) __bf16 bf16x16;
typedef __attribute__((ext_vector_type(8)))  float  floatx8;

// ---------- f32 -> bf16 (round to nearest even) ----------
__device__ __forceinline__ unsigned short f2bf(float f) {
    unsigned int u = __float_as_uint(f);
    u += 0x7FFFu + ((u >> 16) & 1u);
    return (unsigned short)(u >> 16);
}

__global__ __launch_bounds__(256) void cvt_f32_bf16(const float* __restrict__ in,
                                                    unsigned short* __restrict__ out,
                                                    int n) {
    for (int i = blockIdx.x * blockDim.x + threadIdx.x; i < n; i += gridDim.x * blockDim.x)
        out[i] = f2bf(in[i]);
}

// ---------- WMMA fragment load: 16 bf16/lane as two 16B chunks ----------
struct FragU { union { bf16x16 v; uint4 u[2]; }; };
__device__ __forceinline__ bf16x16 load_frag(const unsigned short* p) {
    FragU f;
    f.u[0] = *(const uint4*)p;          // K = kb .. kb+7
    f.u[1] = *(const uint4*)(p + 16);   // K = kb+16 .. kb+23
    return f.v;
}

// ---------- Gate GEMM: scores = X @ G^T + gate_b + 0.1*noise ----------
// Wave tile 32(M) x 64(N): 2x4 WMMA per K-step (A reused 4x, B reused 2x).
// Block = 8 waves arranged 4(M) x 2(N) -> 128 x 128 block tile.
__global__ __launch_bounds__(256) void gate_gemm_wmma(
    const unsigned short* __restrict__ Xb,   // [NTOK, DIM] bf16
    const unsigned short* __restrict__ Gb,   // [NEXP, DIM] bf16
    const float* __restrict__ gate_b,        // [NEXP]
    const float* __restrict__ noise,         // [NTOK, NEXP]
    float* __restrict__ scores)              // [NTOK, NEXP]
{
    const int lane = threadIdx.x & 31;
    const int wid  = threadIdx.x >> 5;
    const int wm   = wid >> 1;                  // 0..3
    const int wn   = wid & 1;                   // 0..1
    const int m0   = blockIdx.y * 128 + wm * 32;
    const int n0   = blockIdx.x * 128 + wn * 64;

    const int lr     = lane & 15;               // M (A) / N (B) within tile
    const int kboff  = (lane >> 4) * 8;         // K phase per half-wave

    floatx8 acc[2][4] = {};

    for (int k = 0; k < DIM; k += 32) {
        bf16x16 A[2], B[4];
        #pragma unroll
        for (int i = 0; i < 2; ++i)
            A[i] = load_frag(Xb + (size_t)(m0 + i * 16 + lr) * DIM + k + kboff);
        #pragma unroll
        for (int j = 0; j < 4; ++j)
            B[j] = load_frag(Gb + (size_t)(n0 + j * 16 + lr) * DIM + k + kboff);
        #pragma unroll
        for (int i = 0; i < 2; ++i)
            #pragma unroll
            for (int j = 0; j < 4; ++j)
                acc[i][j] = __builtin_amdgcn_wmma_f32_16x16x32_bf16(
                    false, A[i], false, B[j], (short)0, acc[i][j], false, false);
    }

    // D layout: lane 0-15 -> N=lane, M=r ; lane 16-31 -> N=lane-16, M=8+r
    const int mphase = (lane >> 4) * 8;
    #pragma unroll
    for (int i = 0; i < 2; ++i) {
        #pragma unroll
        for (int j = 0; j < 4; ++j) {
            const int col  = n0 + j * 16 + lr;
            const float bias = gate_b[col];
            #pragma unroll
            for (int r = 0; r < 8; ++r) {
                const int row = m0 + i * 16 + r + mphase;
                const size_t o = (size_t)row * NEXP + col;
                scores[o] = acc[i][j][r] + bias + NOISE_SCALE * noise[o];
            }
        }
    }
}

// ---------- Softmax + top-1 per token ----------
__global__ __launch_bounds__(256) void softmax_top1(
    const float* __restrict__ scores, int* __restrict__ top_i, float* __restrict__ top_w)
{
    __shared__ float smax[256];
    __shared__ int   sarg[256];
    __shared__ float ssum[256];
    const int n = blockIdx.x, tid = threadIdx.x;
    const float* row = scores + (size_t)n * NEXP;

    float m = -INFINITY; int a = 0;
    for (int e = tid; e < NEXP; e += 256) {
        float v = row[e];
        if (v > m) { m = v; a = e; }
    }
    smax[tid] = m; sarg[tid] = a;
    __syncthreads();
    for (int s = 128; s > 0; s >>= 1) {
        if (tid < s) {
            if (smax[tid + s] > smax[tid] ||
                (smax[tid + s] == smax[tid] && sarg[tid + s] < sarg[tid])) {
                smax[tid] = smax[tid + s]; sarg[tid] = sarg[tid + s];
            }
        }
        __syncthreads();
    }
    const float gm = smax[0];
    float sum = 0.f;
    for (int e = tid; e < NEXP; e += 256) sum += __expf(row[e] - gm);
    ssum[tid] = sum;
    __syncthreads();
    for (int s = 128; s > 0; s >>= 1) {
        if (tid < s) ssum[tid] += ssum[tid + s];
        __syncthreads();
    }
    if (tid == 0) { top_i[n] = sarg[0]; top_w[n] = 1.0f / ssum[0]; }
}

// ---------- Fused expert matvec + projection ----------
// One block (8 waves) per token; one wave per expert output row (coalesced
// 128B weight streaming); projection from LDS-resident y.
__global__ __launch_bounds__(256) void expert_proj(
    const float* __restrict__ x,         // [NTOK, DIM]
    const int*   __restrict__ top_i,     // [NTOK]
    const float* __restrict__ top_w,     // [NTOK]
    const float* __restrict__ expert_w,  // [NEXP, EDIM, DIM]
    const float* __restrict__ expert_b,  // [NEXP, EDIM]
    const float* __restrict__ proj_w,    // [DIM, EDIM]
    const float* __restrict__ proj_b,    // [DIM]
    float* __restrict__ out)             // [NTOK, DIM]
{
    __shared__ float xs[DIM];
    __shared__ float ys[EDIM];
    const int n    = blockIdx.x;
    const int tid  = threadIdx.x;
    const int lane = tid & 31;
    const int w    = tid >> 5;

    const float* xr = x + (size_t)n * DIM;
    for (int d = tid; d < DIM; d += 256) xs[d] = xr[d];

    const int   e  = top_i[n];
    const float tw = top_w[n];
    const float* W = expert_w + (size_t)e * EDIM * DIM;
    __syncthreads();

    #pragma unroll
    for (int i = 0; i < 8; ++i) {
        const int o = i * 8 + w;
        const float* wr = W + (size_t)o * DIM;
        if (i + 1 < 8)
            __builtin_prefetch(W + (size_t)(o + 8) * DIM + lane * 4, 0, 0);
        float s = 0.f;
        #pragma unroll 4
        for (int d = lane; d < DIM; d += 32) s += xs[d] * wr[d];
        #pragma unroll
        for (int off = 16; off > 0; off >>= 1) s += __shfl_xor(s, off, 32);
        if (lane == 0) ys[o] = s + expert_b[(size_t)e * EDIM + o];
    }
    __syncthreads();

    for (int d = tid; d < DIM; d += 256) {
        const float* pr = proj_w + (size_t)d * EDIM;
        float acc = proj_b[d];
        #pragma unroll 16
        for (int o = 0; o < EDIM; ++o) acc += ys[o] * pr[o];
        out[(size_t)n * DIM + d] = acc * tw;
    }
}

// ---------- Launch ----------
extern "C" void kernel_launch(void* const* d_in, const int* in_sizes, int n_in,
                              void* d_out, int out_size, void* d_ws, size_t ws_size,
                              hipStream_t stream) {
    const float* x        = (const float*)d_in[0];
    const float* noise    = (const float*)d_in[1];
    const float* gate_w   = (const float*)d_in[2];
    const float* gate_b   = (const float*)d_in[3];
    const float* expert_w = (const float*)d_in[4];
    const float* expert_b = (const float*)d_in[5];
    const float* proj_w   = (const float*)d_in[6];
    const float* proj_b   = (const float*)d_in[7];
    float* out = (float*)d_out;

    char* ws = (char*)d_ws;
    unsigned short* xb  = (unsigned short*)(ws);                 // 4096*768*2  = 6,291,456
    unsigned short* gb  = (unsigned short*)(ws + 6291456);       // 1536*768*2  = 2,359,296
    float*          sc  = (float*)(ws + 8650752);                // 4096*1536*4 = 25,165,824
    int*            ti  = (int*)(ws + 33816576);                 // 4096*4
    float*          tw  = (float*)(ws + 33832960);               // 4096*4

    cvt_f32_bf16<<<2048, 256, 0, stream>>>(x,      xb, NTOK * DIM);
    cvt_f32_bf16<<<1024, 256, 0, stream>>>(gate_w, gb, NEXP * DIM);

    dim3 g(NEXP / 128, NTOK / 128);                              // 12 x 32
    gate_gemm_wmma<<<g, 256, 0, stream>>>(xb, gb, gate_b, noise, sc);

    softmax_top1<<<NTOK, 256, 0, stream>>>(sc, ti, tw);

    expert_proj<<<NTOK, 256, 0, stream>>>(x, ti, tw, expert_w, expert_b,
                                          proj_w, proj_b, out);
}
